// VariationalEncoder_41016937677104
// MI455X (gfx1250) — compile-verified, matrix operand order checked by reading, not run on hardware
//
#include <hip/hip_runtime.h>

// ---------------------------------------------------------------------------
// Problem constants (match reference)
// ---------------------------------------------------------------------------
constexpr int BC = 4, BL = 4096, BS = 2048, BT = 256, BR = 64, BH = 128;
constexpr int NHEADS = 4, NLAYERS = 2, DHD = 32, BFF = 512;
constexpr float ATT_SCALE = 0.17677669529663687f;   // DH^-0.5
constexpr float EPS_RMS = 1.1920929e-07f;
constexpr size_t CSZ = (size_t)BC * BS;             // total tokens

typedef __attribute__((ext_vector_type(16))) __bf16 v16bf;
typedef __attribute__((ext_vector_type(8)))  float  v8f;
typedef __attribute__((ext_vector_type(8)))  unsigned v8u;

#define DEVFN __device__ __forceinline__

DEVFN unsigned short f2bfbits(float f) {
  unsigned u = __builtin_bit_cast(unsigned, f);
  unsigned r = u + 0x7FFFu + ((u >> 16) & 1u);   // round-to-nearest-even
  return (unsigned short)(r >> 16);
}
DEVFN __bf16 f2bf(float f) {
  unsigned short h = f2bfbits(f);
  return __builtin_bit_cast(__bf16, h);
}
DEVFN float gelu_erf(float x) { return 0.5f * x * (1.0f + erff(x * 0.70710678118654752f)); }

// Build a WMMA 16-bit fragment from two contiguous 16-byte chunks
// (elements 0..7 <- p0[0..7], elements 8..15 <- p1[0..7]); p0/p1 16B aligned.
DEVFN v16bf load_frag(const unsigned short* p0, const unsigned short* p1) {
  uint4 a = *(const uint4*)p0;
  uint4 b = *(const uint4*)p1;
  v8u t;
  t[0] = a.x; t[1] = a.y; t[2] = a.z; t[3] = a.w;
  t[4] = b.x; t[5] = b.y; t[6] = b.z; t[7] = b.w;
  return __builtin_bit_cast(v16bf, t);
}

// ---------------------------------------------------------------------------
// Selection scans
// ---------------------------------------------------------------------------
__global__ void build_sel_kernel(const int* __restrict__ tg, const int* __restrict__ tf,
                                 int* __restrict__ sel, int* __restrict__ tfpos) {
  if (threadIdx.x != 0) return;
  int c = blockIdx.x;
  int cnt = 0;
  for (int i = 0; i < BL && cnt < BS; ++i)
    if (tg[c * BL + i]) sel[c * BS + cnt++] = i;
  int c2 = 0;
  for (int s = 0; s < BS && c2 < BT; ++s)
    if (tf[c * BL + sel[c * BS + s]]) tfpos[c * BT + c2++] = s;
}

__global__ void zero_u16_kernel(unsigned short* __restrict__ p, long long n) {
  long long i = (long long)blockIdx.x * blockDim.x + threadIdx.x;
  if (i < n) p[i] = 0;
}

// u (C,T,R) f32 -> scatter into u_full (C,S,R) bf16 at tf positions
__global__ void scatter_u_kernel(const float* __restrict__ u, const int* __restrict__ tfpos,
                                 unsigned short* __restrict__ u_full) {
  long long i = (long long)blockIdx.x * blockDim.x + threadIdx.x;
  if (i >= (long long)BC * BT * BR) return;
  int c = (int)(i / (BT * BR));
  int rem = (int)(i % (BT * BR));
  int t = rem / BR, r = rem % BR;
  int pos = tfpos[c * BT + t];
  u_full[((size_t)c * BS + pos) * BR + r] = f2bfbits(u[i]);
}

__global__ void f32_to_bf16_kernel(const float* __restrict__ src,
                                   unsigned short* __restrict__ dst, long long n) {
  long long i = (long long)blockIdx.x * blockDim.x + threadIdx.x;
  if (i < n) dst[i] = f2bfbits(src[i]);
}

// weight transpose-convert: src (K,N) f32 row-major -> dst (N,K) bf16 row-major
// batched over blockIdx.y.
__global__ void wT_bf16_kernel(const float* __restrict__ src, unsigned short* __restrict__ dst,
                               int K, int N) {
  long long i = (long long)blockIdx.x * blockDim.x + threadIdx.x;
  if (i >= (long long)K * N) return;
  int k = (int)(i / N), n = (int)(i % N);
  const float* s = src + (size_t)blockIdx.y * K * N;
  unsigned short* d = dst + (size_t)blockIdx.y * K * N;
  d[(size_t)n * K + k] = f2bfbits(s[i]);
}

// g[row,j] = gelu(x[c, sel[row]] * w_e1[j] + b_e1[j]) -> (C*S, H) bf16
__global__ void expr_gate_kernel(const float* __restrict__ x, const int* __restrict__ sel,
                                 const float* __restrict__ w_e1, const float* __restrict__ b_e1,
                                 unsigned short* __restrict__ g) {
  long long i = (long long)blockIdx.x * blockDim.x + threadIdx.x;
  if (i >= (long long)BC * BS * BH) return;
  int row = (int)(i / BH);
  int j = (int)(i % BH);
  int c = row / BS;
  float xv = x[c * BL + sel[row]];
  g[i] = f2bfbits(gelu_erf(xv * w_e1[j] + b_e1[j]));
}

// ---------------------------------------------------------------------------
// Generic WMMA GEMM:  Out = op(A_bf16 @ B + bias) [+ residual_f32]
//   A : (M,K) bf16 row-major
//   BT: (N,K) bf16 row-major (i.e. B transposed -> contiguous fragments)
// One wave computes a 16x32 strip (two 16x16 tiles share the A fragment).
// Epilogues: +bias, GELU, f32 out, bf16 out, bf16 transposed out (N,M).
// ---------------------------------------------------------------------------
enum { GF_GELU = 1, GF_OUTBF16 = 2, GF_OUTTRANS = 4 };

__global__ __launch_bounds__(128)
void gemm_kernel(const unsigned short* __restrict__ A, long long sA,
                 const unsigned short* __restrict__ BTm, long long sB,
                 void* __restrict__ OutP, long long sO,
                 const float* __restrict__ bias,
                 const float* __restrict__ resid, long long sR,
                 int M, int N, int K, int flags) {
  int lane = threadIdx.x & 31;
  int wave = threadIdx.x >> 5;
  int pairsN = N >> 5, tilesM = M >> 4;
  int tile = blockIdx.x * 4 + wave;
  if (tile >= tilesM * pairsN) return;
  int tm = tile / pairsN, tp = tile - tm * pairsN;
  int b = blockIdx.y;
  const unsigned short* Ab = A + (size_t)b * sA;
  const unsigned short* Bb = BTm + (size_t)b * sB;
  int half = lane >> 4, l16 = lane & 15;
  int kb = half * 8;
  int mrow = tm * 16 + l16;
  int n0 = tp * 32 + l16;
  int n1 = n0 + 16;
  const unsigned short* ap  = Ab + (size_t)mrow * K + kb;
  const unsigned short* bp0 = Bb + (size_t)n0 * K + kb;
  const unsigned short* bp1 = Bb + (size_t)n1 * K + kb;
  v8f acc0 = {}, acc1 = {};
  for (int k0 = 0; k0 < K; k0 += 32) {
    v16bf af  = load_frag(ap + k0, ap + k0 + 16);
    v16bf bf0 = load_frag(bp0 + k0, bp0 + k0 + 16);
    v16bf bf1 = load_frag(bp1 + k0, bp1 + k0 + 16);
    acc0 = __builtin_amdgcn_wmma_f32_16x16x32_bf16(false, af, false, bf0, (short)0, acc0, false, false);
    acc1 = __builtin_amdgcn_wmma_f32_16x16x32_bf16(false, af, false, bf1, (short)0, acc1, false, false);
  }
  int mb = half * 8;
#pragma unroll
  for (int t = 0; t < 2; ++t) {
    v8f acc = t ? acc1 : acc0;
    int ncol = t ? n1 : n0;
    float bv = bias ? bias[ncol] : 0.f;
    if (flags & GF_OUTTRANS) {
      // contiguous packed bf16 store along M
      unsigned short tmp[8];
#pragma unroll
      for (int r = 0; r < 8; ++r) {
        float v = acc[r] + bv;
        if (flags & GF_GELU) v = gelu_erf(v);
        tmp[r] = f2bfbits(v);
      }
      uint4 pk;
      pk.x = (unsigned)tmp[0] | ((unsigned)tmp[1] << 16);
      pk.y = (unsigned)tmp[2] | ((unsigned)tmp[3] << 16);
      pk.z = (unsigned)tmp[4] | ((unsigned)tmp[5] << 16);
      pk.w = (unsigned)tmp[6] | ((unsigned)tmp[7] << 16);
      *(uint4*)(((unsigned short*)OutP) + (size_t)b * sO + (size_t)ncol * M + tm * 16 + mb) = pk;
    } else {
#pragma unroll
      for (int r = 0; r < 8; ++r) {
        int row = tm * 16 + mb + r;
        float v = acc[r] + bv;
        if (flags & GF_GELU) v = gelu_erf(v);
        size_t oidx = (size_t)row * N + ncol;
        if (resid) v += resid[(size_t)b * sR + oidx];
        if (flags & GF_OUTBF16)
          ((unsigned short*)OutP)[(size_t)b * sO + oidx] = f2bfbits(v);
        else
          ((float*)OutP)[(size_t)b * sO + oidx] = v;
      }
    }
  }
}

// ---------------------------------------------------------------------------
// ew finalize: ew += I, row-normalize by max(rowsum,1e-8), log(max(.,1e-8))
// ---------------------------------------------------------------------------
__global__ void ew_finalize_kernel(float* __restrict__ ew) {
  int row = blockIdx.x;            // c*BS + s
  int s = row & (BS - 1);
  float* p = ew + (size_t)row * BS;
  int lane = threadIdx.x;
  float sum = 0.f;
  for (int j = lane; j < BS; j += 32) {
    float v = p[j] + (j == s ? 1.0f : 0.0f);
    p[j] = v;
    sum += v;
  }
  for (int m = 1; m < 32; m <<= 1) sum += __shfl_xor(sum, m, 32);
  float inv = 1.0f / fmaxf(sum, 1e-8f);
  for (int j = lane; j < BS; j += 32)
    p[j] = logf(fmaxf(p[j] * inv, 1e-8f));
}

// ---------------------------------------------------------------------------
// RMSNorm: one wave per row of H=128, bf16 output (GEMM A operand).
// ---------------------------------------------------------------------------
__global__ void rmsnorm_kernel(const float* __restrict__ X, const float* __restrict__ W,
                               unsigned short* __restrict__ O) {
  int row = blockIdx.x;
  int lane = threadIdx.x;
  const float* xp = X + (size_t)row * BH;
  float v0 = xp[lane], v1 = xp[lane + 32], v2 = xp[lane + 64], v3 = xp[lane + 96];
  float s = v0 * v0 + v1 * v1 + v2 * v2 + v3 * v3;
  for (int m = 1; m < 32; m <<= 1) s += __shfl_xor(s, m, 32);
  float inv = rsqrtf(s * (1.0f / BH) + EPS_RMS);
  unsigned short* op = O + (size_t)row * BH;
  op[lane]      = f2bfbits(v0 * inv * W[lane]);
  op[lane + 32] = f2bfbits(v1 * inv * W[lane + 32]);
  op[lane + 64] = f2bfbits(v2 * inv * W[lane + 64]);
  op[lane + 96] = f2bfbits(v3 * inv * W[lane + 96]);
}

// ---------------------------------------------------------------------------
// Flash attention, one wave per (c, head, 16-query tile). DH=32 == WMMA K.
// Q,K: (C,S,H) bf16.  V: head-transposed (H, C*S) bf16 -> contiguous frags.
// Per 32-key block: 2 score WMMAs, online softmax, 2 P@V WMMAs.
// Output: (C,S,H) bf16 (A operand of the O-projection GEMM).
// ---------------------------------------------------------------------------
__global__ __launch_bounds__(32)
void attn_kernel(const unsigned short* __restrict__ Q,
                 const unsigned short* __restrict__ Kmat,
                 const unsigned short* __restrict__ VT,
                 const float* __restrict__ LB,
                 unsigned short* __restrict__ O) {
  int qt = blockIdx.x, head = blockIdx.y, c = blockIdx.z;
  int lane = threadIdx.x & 31;
  int half = lane >> 4, l16 = lane & 15;
  int kb = half * 8, mb = half * 8;
  __shared__ float pt[16][33];

  const unsigned short* qp =
      Q + ((size_t)(c * BS + qt * 16 + l16)) * BH + head * DHD + kb;
  v16bf qa = load_frag(qp, qp + 16);

  const unsigned short* v0base = VT + (size_t)(head * DHD + l16) * CSZ + (size_t)c * BS + kb;
  const unsigned short* v1base = VT + (size_t)(head * DHD + 16 + l16) * CSZ + (size_t)c * BS + kb;

  float mstat[8], lstat[8];
  v8f o0 = {}, o1 = {};
#pragma unroll
  for (int r = 0; r < 8; ++r) { mstat[r] = -3.0e38f; lstat[r] = 0.f; }

  for (int kblk = 0; kblk < BS / 32; ++kblk) {
    const unsigned short* kp =
        Kmat + ((size_t)(c * BS + kblk * 32 + l16)) * BH + head * DHD + kb;
    __builtin_prefetch(kp + (size_t)32 * BH, 0, 1);       // next key block
    __builtin_prefetch(v0base + kblk * 32 + 32, 0, 1);    // next value block
    v16bf kf0 = load_frag(kp, kp + 16);
    v16bf kf1 = load_frag(kp + (size_t)16 * BH, kp + (size_t)16 * BH + 16);
    v8f s0 = {}, s1 = {};
    s0 = __builtin_amdgcn_wmma_f32_16x16x32_bf16(false, qa, false, kf0, (short)0, s0, false, false);
    s1 = __builtin_amdgcn_wmma_f32_16x16x32_bf16(false, qa, false, kf1, (short)0, s1, false, false);

    int n0 = kblk * 32 + l16;
#pragma unroll
    for (int r = 0; r < 8; ++r) {
      int qrow = qt * 16 + mb + r;
      const float* lbp = LB + ((size_t)c * BS + qrow) * BS;
      float v0 = s0[r] * ATT_SCALE + lbp[n0];
      float v1 = s1[r] * ATT_SCALE + lbp[n0 + 16];
      float t = fmaxf(v0, v1);
      for (int m = 1; m < 16; m <<= 1) t = fmaxf(t, __shfl_xor(t, m, 16));
      float mnew = fmaxf(mstat[r], t);
      float p0 = expf(v0 - mnew);
      float p1 = expf(v1 - mnew);
      float rs = p0 + p1;
      for (int m = 1; m < 16; m <<= 1) rs += __shfl_xor(rs, m, 16);
      float corr = expf(mstat[r] - mnew);
      lstat[r] = lstat[r] * corr + rs;
      mstat[r] = mnew;
      o0[r] *= corr; o1[r] *= corr;
      pt[mb + r][l16] = p0;
      pt[mb + r][16 + l16] = p1;
    }
    __syncthreads();
    v16bf pa;
#pragma unroll
    for (int i = 0; i < 8; ++i) pa[i] = f2bf(pt[l16][kb + i]);
#pragma unroll
    for (int i = 0; i < 8; ++i) pa[8 + i] = f2bf(pt[l16][16 + kb + i]);
    __syncthreads();
    const unsigned short* vp0 = v0base + kblk * 32;
    const unsigned short* vp1 = v1base + kblk * 32;
    v16bf vf0 = load_frag(vp0, vp0 + 16);
    v16bf vf1 = load_frag(vp1, vp1 + 16);
    o0 = __builtin_amdgcn_wmma_f32_16x16x32_bf16(false, pa, false, vf0, (short)0, o0, false, false);
    o1 = __builtin_amdgcn_wmma_f32_16x16x32_bf16(false, pa, false, vf1, (short)0, o1, false, false);
  }
#pragma unroll
  for (int r = 0; r < 8; ++r) {
    int qrow = qt * 16 + mb + r;
    float inv = 1.0f / lstat[r];
    unsigned short* op = O + ((size_t)(c * BS + qrow)) * BH + head * DHD;
    op[l16]      = f2bfbits(o0[r] * inv);
    op[16 + l16] = f2bfbits(o1[r] * inv);
  }
}

// ---------------------------------------------------------------------------
// Final head: rmsnorm(h, nw) @ zw + zb -> mu, softplus(.)+1e-6 -> sigma.
// ---------------------------------------------------------------------------
__global__ void head_kernel(const float* __restrict__ Hbuf, const float* __restrict__ nw,
                            const float* __restrict__ zw, const float* __restrict__ zb,
                            float* __restrict__ out) {
  int row = blockIdx.x;
  int lane = threadIdx.x;
  const float* xp = Hbuf + (size_t)row * BH;
  float vals[4];
  float s = 0.f;
#pragma unroll
  for (int j = 0; j < 4; ++j) { vals[j] = xp[lane + 32 * j]; s += vals[j] * vals[j]; }
  for (int m = 1; m < 32; m <<= 1) s += __shfl_xor(s, m, 32);
  float inv = rsqrtf(s * (1.0f / BH) + EPS_RMS);
  float a0 = 0.f, a1 = 0.f;
#pragma unroll
  for (int j = 0; j < 4; ++j) {
    int idx = lane + 32 * j;
    float xn = vals[j] * inv * nw[idx];
    a0 += xn * zw[idx * 2];
    a1 += xn * zw[idx * 2 + 1];
  }
  for (int m = 1; m < 32; m <<= 1) { a0 += __shfl_xor(a0, m, 32); a1 += __shfl_xor(a1, m, 32); }
  if (lane == 0) {
    out[row] = a0 + zb[0];
    float z = a1 + zb[1];
    float sp = (z > 20.f) ? z : log1pf(expf(z));
    out[BC * BS + row] = sp + 1e-6f;
  }
}

// ---------------------------------------------------------------------------
// Host-side orchestration
// ---------------------------------------------------------------------------
static inline void launch_gemm(hipStream_t st, const unsigned short* A, long long sA,
                               const unsigned short* BTm, long long sB,
                               void* Out, long long sO,
                               const float* bias, const float* resid, long long sR,
                               int M, int N, int K, int flags, int batches) {
  int tiles = (M / 16) * (N / 32);
  int blocks = (tiles + 3) / 4;
  gemm_kernel<<<dim3(blocks, batches), 128, 0, st>>>(A, sA, BTm, sB, Out, sO,
                                                     bias, resid, sR, M, N, K, flags);
}

static inline void launch_wT(hipStream_t st, const float* src, unsigned short* dst,
                             int K, int N, int batches) {
  long long n = (long long)K * N;
  wT_bf16_kernel<<<dim3((unsigned)((n + 255) / 256), batches), 256, 0, st>>>(src, dst, K, N);
}

extern "C" void kernel_launch(void* const* d_in, const int* in_sizes, int n_in,
                              void* d_out, int out_size, void* d_ws, size_t ws_size,
                              hipStream_t stream) {
  (void)in_sizes; (void)n_in; (void)out_size; (void)ws_size;
  const float* x     = (const float*)d_in[0];
  const int*   btf   = (const int*)d_in[1];
  const int*   btg   = (const int*)d_in[2];
  const float* u     = (const float*)d_in[3];
  const float* v     = (const float*)d_in[4];
  const float* w_e1  = (const float*)d_in[5];
  const float* b_e1  = (const float*)d_in[6];
  const float* w_e2  = (const float*)d_in[7];
  const float* b_e2  = (const float*)d_in[8];
  const float* n1w   = (const float*)d_in[9];
  const float* qw    = (const float*)d_in[10];
  const float* qbias = (const float*)d_in[11];
  const float* kw    = (const float*)d_in[12];
  const float* kbias = (const float*)d_in[13];
  const float* vw    = (const float*)d_in[14];
  const float* vbias = (const float*)d_in[15];
  const float* ow    = (const float*)d_in[16];
  const float* obias = (const float*)d_in[17];
  const float* n2w   = (const float*)d_in[18];
  const float* f1w   = (const float*)d_in[19];
  const float* f1b   = (const float*)d_in[20];
  const float* f2w   = (const float*)d_in[21];
  const float* f2b   = (const float*)d_in[22];
  const float* nw    = (const float*)d_in[23];
  const float* zw    = (const float*)d_in[24];
  const float* zb    = (const float*)d_in[25];

  char* wp = (char*)d_ws;
  auto alloc = [&](size_t bytes) -> void* {
    void* p = (void*)wp;
    wp += (bytes + 255) & ~(size_t)255;
    return p;
  };
  const size_t CSH = CSZ * BH;
  int*            sel    = (int*)alloc((size_t)BC * BS * 4);
  int*            tfpos  = (int*)alloc((size_t)BC * BT * 4);
  unsigned short* u_full = (unsigned short*)alloc((size_t)BC * BS * BR * 2);
  unsigned short* vbf    = (unsigned short*)alloc((size_t)BC * BS * BR * 2);
  float*          ew     = (float*)alloc((size_t)BC * BS * BS * 4);   // 64 MB -> log_bias
  unsigned short* gbuf   = (unsigned short*)alloc(CSH * 2);
  float*          hbuf   = (float*)alloc(CSH * 4);
  unsigned short* xnb    = (unsigned short*)alloc(CSH * 2);
  unsigned short* qbuf   = (unsigned short*)alloc(CSH * 2);
  unsigned short* kbuf   = (unsigned short*)alloc(CSH * 2);
  unsigned short* vtbuf  = (unsigned short*)alloc(CSH * 2);           // (H, C*S)
  unsigned short* obuf   = (unsigned short*)alloc(CSH * 2);
  unsigned short* t1     = (unsigned short*)alloc((size_t)BC * BS * BFF * 2);
  unsigned short* we2T   = (unsigned short*)alloc((size_t)BH * BH * 2);
  unsigned short* qwT    = (unsigned short*)alloc((size_t)NLAYERS * BH * BH * 2);
  unsigned short* kwT    = (unsigned short*)alloc((size_t)NLAYERS * BH * BH * 2);
  unsigned short* vwT    = (unsigned short*)alloc((size_t)NLAYERS * BH * BH * 2);
  unsigned short* owT    = (unsigned short*)alloc((size_t)NLAYERS * BH * BH * 2);
  unsigned short* f1wT   = (unsigned short*)alloc((size_t)NLAYERS * BH * BFF * 2);
  unsigned short* f2wT   = (unsigned short*)alloc((size_t)NLAYERS * BFF * BH * 2);

  // ---- preamble ----
  build_sel_kernel<<<BC, 32, 0, stream>>>(btg, btf, sel, tfpos);
  {
    long long n = (long long)BC * BS * BR;
    zero_u16_kernel<<<(unsigned)((n + 255) / 256), 256, 0, stream>>>(u_full, n);
  }
  {
    long long n = (long long)BC * BT * BR;
    scatter_u_kernel<<<(unsigned)((n + 255) / 256), 256, 0, stream>>>(u, tfpos, u_full);
  }
  {
    long long n = (long long)BC * BS * BR;   // v (C,S,R) is already (N,K) for the ew GEMM
    f32_to_bf16_kernel<<<(unsigned)((n + 255) / 256), 256, 0, stream>>>(v, vbf, n);
  }
  launch_wT(stream, w_e2, we2T, BH, BH, 1);
  launch_wT(stream, qw,  qwT,  BH, BH, NLAYERS);
  launch_wT(stream, kw,  kwT,  BH, BH, NLAYERS);
  launch_wT(stream, vw,  vwT,  BH, BH, NLAYERS);
  launch_wT(stream, ow,  owT,  BH, BH, NLAYERS);
  launch_wT(stream, f1w, f1wT, BH, BFF, NLAYERS);
  launch_wT(stream, f2w, f2wT, BFF, BH, NLAYERS);

  // ew = u_full @ v^T (batched over C), then +I, row-normalize, log
  launch_gemm(stream, u_full, (long long)BS * BR, vbf, (long long)BS * BR,
              ew, (long long)BS * BS, nullptr, nullptr, 0,
              BS, BS, BR, 0, BC);
  ew_finalize_kernel<<<BC * BS, 32, 0, stream>>>(ew);

  // expr projection: gbuf = gelu(x_sel * w_e1 + b_e1); h = gbuf @ w_e2 + b_e2
  {
    long long n = (long long)CSH;
    expr_gate_kernel<<<(unsigned)((n + 255) / 256), 256, 0, stream>>>(x, sel, w_e1, b_e1, gbuf);
  }
  launch_gemm(stream, gbuf, 0, we2T, 0, hbuf, 0, b_e2, nullptr, 0,
              BC * BS, BH, BH, 0, 1);

  // ---- transformer layers ----
  for (int i = 0; i < NLAYERS; ++i) {
    rmsnorm_kernel<<<BC * BS, 32, 0, stream>>>(hbuf, n1w + i * BH, xnb);
    launch_gemm(stream, xnb, 0, qwT + (size_t)i * BH * BH, 0, qbuf, 0,
                qbias + i * BH, nullptr, 0, BC * BS, BH, BH, GF_OUTBF16, 1);
    launch_gemm(stream, xnb, 0, kwT + (size_t)i * BH * BH, 0, kbuf, 0,
                kbias + i * BH, nullptr, 0, BC * BS, BH, BH, GF_OUTBF16, 1);
    launch_gemm(stream, xnb, 0, vwT + (size_t)i * BH * BH, 0, vtbuf, 0,
                vbias + i * BH, nullptr, 0, BC * BS, BH, BH, GF_OUTBF16 | GF_OUTTRANS, 1);
    attn_kernel<<<dim3(BS / 16, NHEADS, BC), 32, 0, stream>>>(qbuf, kbuf, vtbuf, ew, obuf);
    launch_gemm(stream, obuf, 0, owT + (size_t)i * BH * BH, 0, hbuf, 0,
                obias + i * BH, hbuf, 0, BC * BS, BH, BH, 0, 1);
    rmsnorm_kernel<<<BC * BS, 32, 0, stream>>>(hbuf, n2w + i * BH, xnb);
    launch_gemm(stream, xnb, 0, f1wT + (size_t)i * BH * BFF, 0, t1, 0,
                f1b + i * BFF, nullptr, 0, BC * BS, BFF, BH, GF_GELU | GF_OUTBF16, 1);
    launch_gemm(stream, t1, 0, f2wT + (size_t)i * BFF * BH, 0, hbuf, 0,
                f2b + i * BH, hbuf, 0, BC * BS, BH, BFF, 0, 1);
  }

  // ---- output head ----
  head_kernel<<<BC * BS, 32, 0, stream>>>(hbuf, nw, zw, zb, (float*)d_out);
}
